// QFCModel_65481071408380
// MI455X (gfx1250) — matrix operand order repaired and checked by reading, not compiled, not run on hardware
//
#include <hip/hip_runtime.h>

typedef __attribute__((ext_vector_type(16))) _Float16 v16h;
typedef __attribute__((ext_vector_type(8)))  float    v8f;

// ---------------------------------------------------------------------------
// Kernel 0: fold the fixed parameterized circuit (3 layers of RZ/RX/RY per
// wire + CNOT ring) into a real 32x32 matrix W acting on [Re psi ; Im psi].
// Thread j evolves basis state |j> -> column j of the 16x16 complex unitary U.
// W = [[Ur^T, Ui^T], [-Ui^T, Ur^T]]  (row-major, out = in @ W).
// Wire w corresponds to bit (3-w) of the basis index (MSB-first axes).
// ---------------------------------------------------------------------------
__global__ void qfc_build_w(const float* __restrict__ params, float* __restrict__ W) {
  const int j = threadIdx.x;
  if (j >= 16) return;
  float sr[16], si[16];
#pragma unroll
  for (int k = 0; k < 16; ++k) { sr[k] = (k == j) ? 1.f : 0.f; si[k] = 0.f; }

  for (int l = 0; l < 3; ++l) {
#pragma unroll
    for (int w = 0; w < 4; ++w) {
      const int bit = 8 >> w;
      float t, ch, sh;
      // RZ(t): diag(e^{-it/2}, e^{+it/2})
      t = params[l * 12 + w * 3 + 0] * 0.5f; ch = cosf(t); sh = sinf(t);
#pragma unroll
      for (int k = 0; k < 16; ++k) {
        float r = sr[k], im = si[k];
        if (k & bit) { sr[k] = r * ch - im * sh; si[k] = im * ch + r * sh; }
        else         { sr[k] = r * ch + im * sh; si[k] = im * ch - r * sh; }
      }
      // RX(t): [[c, -i s], [-i s, c]]
      t = params[l * 12 + w * 3 + 1] * 0.5f; ch = cosf(t); sh = sinf(t);
#pragma unroll
      for (int k = 0; k < 16; ++k) {
        if (k & bit) continue;
        const int k1 = k | bit;
        float r0 = sr[k], i0 = si[k], r1 = sr[k1], i1 = si[k1];
        sr[k]  = ch * r0 + sh * i1;  si[k]  = ch * i0 - sh * r1;
        sr[k1] = ch * r1 + sh * i0;  si[k1] = ch * i1 - sh * r0;
      }
      // RY(t): [[c, -s], [s, c]]
      t = params[l * 12 + w * 3 + 2] * 0.5f; ch = cosf(t); sh = sinf(t);
#pragma unroll
      for (int k = 0; k < 16; ++k) {
        if (k & bit) continue;
        const int k1 = k | bit;
        float r0 = sr[k], i0 = si[k], r1 = sr[k1], i1 = si[k1];
        sr[k]  = ch * r0 - sh * r1;  si[k]  = ch * i0 - sh * i1;
        sr[k1] = sh * r0 + ch * r1;  si[k1] = sh * i0 + ch * i1;
      }
    }
    // CNOT ring: (0->1),(1->2),(2->3),(3->0)
#pragma unroll
    for (int w = 0; w < 4; ++w) {
      const int cb = 8 >> w;
      const int tb = 8 >> ((w + 1) & 3);
#pragma unroll
      for (int k = 0; k < 16; ++k) {
        if ((k & cb) && !(k & tb)) {
          const int k1 = k | tb;
          float tr = sr[k]; sr[k] = sr[k1]; sr[k1] = tr;
          float ti = si[k]; si[k] = si[k1]; si[k1] = ti;
        }
      }
    }
  }
#pragma unroll
  for (int n = 0; n < 16; ++n) {
    W[j * 32 + n]             =  sr[n];
    W[j * 32 + 16 + n]        =  si[n];
    W[(16 + j) * 32 + n]      = -si[n];
    W[(16 + j) * 32 + 16 + n] =  sr[n];
  }
}

// ---------------------------------------------------------------------------
// Kernel 1: main streaming kernel. 256 threads = 8 waves; each wave owns a
// tile of 16 batch items. Per item: coalesced float4 streaming of 576 floats
// with wire-segmented accumulation, butterfly reduce -> 4 encoding sums.
// Then per-lane product-state amplitudes -> f16 A fragment, two
// v_wmma_f32_16x16x32_f16 against the fixed 32x32 W (B fragments), |.|^2,
// sign-weighted butterflies -> <Z_w>, LDS transpose -> coalesced store,
// per-block BN partial sums to fixed slots (deterministic; no atomics).
// ---------------------------------------------------------------------------
__global__ void __launch_bounds__(256) qfc_main(const float* __restrict__ x,
                                                const float* __restrict__ W,
                                                float* __restrict__ exps,
                                                float* __restrict__ partial) {
  __shared__ float ldsT[8][64];
  __shared__ float ldsS[8][8];

  const int tid  = threadIdx.x;
  const int lane = tid & 31;
  const int wv   = tid >> 5;
  const int tile = blockIdx.x * 8 + wv;
  const int item0 = tile * 16;

  // ---- B fragments (32x16 f16 layout: lanes 0-15 hold K=0..7,16..23, lanes
  //      16-31 hold K=8..15,24..31; N = lane&15) ----
  v16h b0, b1;
  {
    const int colA  = lane & 15;
    const int rbase = (lane < 16) ? 0 : 8;
#pragma unroll
    for (int e = 0; e < 16; ++e) {
      b0[e] = (_Float16)W[(rbase + e + (e & 8)) * 32 + colA];
      b1[e] = (_Float16)W[(rbase + e + (e & 8)) * 32 + 16 + colA];
    }
  }

  // ---- encoding row-block sums (wire w = float4 indices [36w, 36w+36)) ----
  float e0 = 0.f, e1 = 0.f, e2 = 0.f, e3 = 0.f;
  const float4* x4 = (const float4*)x;
  for (int i = 0; i < 16; ++i) {
    const float4* p = x4 + (size_t)(item0 + i) * 144;
    float a0, a1 = 0.f, a2 = 0.f, a3 = 0.f;
    float4 v; float s;
    v = p[lane];       a0 = v.x + v.y + v.z + v.w;
    v = p[lane + 32];  s = v.x + v.y + v.z + v.w; if (lane < 4)  a0 += s; else a1 += s;
    v = p[lane + 64];  s = v.x + v.y + v.z + v.w; if (lane < 8)  a1 += s; else a2 += s;
    v = p[lane + 96];  s = v.x + v.y + v.z + v.w; if (lane < 12) a2 += s; else a3 += s;
    if (lane < 16) { v = p[lane + 128]; a3 += v.x + v.y + v.z + v.w; }
#pragma unroll
    for (int m = 1; m <= 16; m <<= 1) {
      a0 += __shfl_xor(a0, m, 32);
      a1 += __shfl_xor(a1, m, 32);
      a2 += __shfl_xor(a2, m, 32);
      a3 += __shfl_xor(a3, m, 32);
    }
    if ((lane & 15) == i) { e0 = a0; e1 = a1; e2 = a2; e3 = a3; }
  }

  // half-angle: enc = sum/36, enc/2 = sum/72
  const float kInv = 1.f / 72.f;
  float c0, s0, c1, s1, c2, s2, c3, s3;
  __sincosf(e0 * kInv, &s0, &c0);
  __sincosf(e1 * kInv, &s1, &c1);
  __sincosf(e2 * kInv, &s2, &c2);
  __sincosf(e3 * kInv, &s3, &c3);

  // ---- A fragment (16x32 f16 layout): item M = lane&15.
  // lanes 0-15:  elems 0-7 = K0..7 (Re psi[0..7]),  elems 8-15 = K16..23 (Im psi[0..7])
  // lanes 16-31: elems 0-7 = K8..15 (Re psi[8..15]), elems 8-15 = K24..31 (Im psi[8..15])
  v16h afrag;
  {
    const int base = (lane < 16) ? 0 : 8;
#pragma unroll
    for (int j = 0; j < 8; ++j) {
      const int k = base + j;
      float m = ((k & 8) ? s0 : c0) * ((k & 4) ? s1 : c1) *
                ((k & 2) ? s2 : c2) * ((k & 1) ? s3 : c3);
      const int pc = __builtin_popcount(k) & 3;   // psi_k = (-i)^pc * m
      float pr = (pc == 0) ? m : ((pc == 2) ? -m : 0.f);
      float pi = (pc == 1) ? -m : ((pc == 3) ? m : 0.f);
      afrag[j]     = (_Float16)pr;
      afrag[8 + j] = (_Float16)pi;
    }
  }

  // ---- WMMA: out(16x32) = psi(16x32) @ W(32x32), split over two N-tiles ----
  v8f cz0 = {};
  v8f cz1 = {};
  v8f d0 = __builtin_amdgcn_wmma_f32_16x16x32_f16(false, afrag, false, b0,
                                                  (short)0, cz0, false, false);
  v8f d1 = __builtin_amdgcn_wmma_f32_16x16x32_f16(false, afrag, false, b1,
                                                  (short)0, cz1, false, false);

  // probs[m][n] = Re^2 + Im^2 (d0 = real cols 0..15, d1 = imag cols 16..31)
  v8f pb;
#pragma unroll
  for (int j = 0; j < 8; ++j) pb[j] = d0[j] * d0[j] + d1[j] * d1[j];

  // ---- <Z_w>: sign-weighted sums over basis index n (= lane&15) ----
  const int n = lane & 15;
  float ex[4][8];
#pragma unroll
  for (int w = 0; w < 4; ++w) {
    const float sgn = ((n >> (3 - w)) & 1) ? -1.f : 1.f;
#pragma unroll
    for (int j = 0; j < 8; ++j) {
      float v = pb[j] * sgn;
      v += __shfl_xor(v, 1, 32);
      v += __shfl_xor(v, 2, 32);
      v += __shfl_xor(v, 4, 32);
      v += __shfl_xor(v, 8, 32);
      ex[w][j] = v;   // items: lanes 0-15 -> j, lanes 16-31 -> 8+j
    }
  }

  // ---- LDS transpose -> coalesced 64-float store per tile ----
  const int g = lane >> 4;
  if ((lane & 15) == 0) {
#pragma unroll
    for (int j = 0; j < 8; ++j)
#pragma unroll
      for (int w = 0; w < 4; ++w)
        ldsT[wv][g * 32 + j * 4 + w] = ex[w][j];
  }
  __syncthreads();
  const float o0 = ldsT[wv][lane];
  const float o1 = ldsT[wv][lane + 32];
  exps[item0 * 4 + lane]      = o0;
  exps[item0 * 4 + 32 + lane] = o1;

  // ---- per-block BatchNorm partials (sum, sumsq per wire) ----
  float sm = o0 + o1;
  float sq = o0 * o0 + o1 * o1;
#pragma unroll
  for (int m = 4; m <= 16; m <<= 1) {   // lanes sharing (lane&3) == wire
    sm += __shfl_xor(sm, m, 32);
    sq += __shfl_xor(sq, m, 32);
  }
  if (lane < 4) { ldsS[wv][lane] = sm; ldsS[wv][4 + lane] = sq; }
  __syncthreads();
  if (tid < 8) {
    float a = 0.f;
#pragma unroll
    for (int k = 0; k < 8; ++k) a += ldsS[k][tid];
    partial[blockIdx.x * 8 + tid] = a;   // [0..3]=sum, [4..7]=sumsq
  }
}

// ---------------------------------------------------------------------------
// Kernel 2: deterministic final reduction of per-block partials -> stats[8].
// ---------------------------------------------------------------------------
__global__ void qfc_reduce(const float* __restrict__ partial,
                           float* __restrict__ stats, int nblocks) {
  const int t = threadIdx.x;
  if (t < 8) {
    float a = 0.f;
    for (int b = 0; b < nblocks; ++b) a += partial[b * 8 + t];
    stats[t] = a;
  }
}

// ---------------------------------------------------------------------------
// Kernel 3: BatchNorm (training mode, biased variance) normalize.
// ---------------------------------------------------------------------------
__global__ void qfc_norm(const float* __restrict__ exps, const float* __restrict__ stats,
                         const float* __restrict__ gamma, const float* __restrict__ beta,
                         float* __restrict__ out, int total, float invB) {
  const int idx = blockIdx.x * blockDim.x + threadIdx.x;
  if (idx >= total) return;
  const int w = idx & 3;
  const float mean = stats[w] * invB;
  const float var  = stats[4 + w] * invB - mean * mean;
  out[idx] = gamma[w] * (exps[idx] - mean) * rsqrtf(var + 1e-5f) + beta[w];
}

// ---------------------------------------------------------------------------
extern "C" void kernel_launch(void* const* d_in, const int* in_sizes, int n_in,
                              void* d_out, int out_size, void* d_ws, size_t ws_size,
                              hipStream_t stream) {
  const float* x      = (const float*)d_in[0];   // (B,1,24,24)
  const float* params = (const float*)d_in[1];   // (3,4,3)
  const float* gamma  = (const float*)d_in[2];   // (4,)
  const float* beta   = (const float*)d_in[3];   // (4,)
  float* out = (float*)d_out;

  const int B = in_sizes[0] / 576;               // 65536
  const int nblocks = B / 128;                   // 512 (8 waves x 16 items each)

  char* ws = (char*)d_ws;
  float* W       = (float*)(ws);                           // 1024 floats (4 KB)
  float* partial = (float*)(ws + 4096);                    // nblocks*8 floats
  float* stats   = (float*)(ws + 4096 + 16384);            // 8 floats
  float* exps    = (float*)(ws + 4096 + 16384 + 256);      // B*4 floats

  qfc_build_w<<<1, 32, 0, stream>>>(params, W);
  qfc_main<<<nblocks, 256, 0, stream>>>(x, W, exps, partial);
  qfc_reduce<<<1, 64, 0, stream>>>(partial, stats, nblocks);

  const int total = B * 4;
  qfc_norm<<<(total + 255) / 256, 256, 0, stream>>>(exps, stats, gamma, beta,
                                                    out, total, 1.f / (float)B);
}
